// PatchedDeepseekV3NaiveMoe_2336462209361
// MI455X (gfx1250) — compile-verified, matrix operand order checked by reading, not run on hardware
//
#include <hip/hip_runtime.h>
#include <math.h>

// Problem constants (from reference)
#define E_   32
#define T_   4096
#define K_   4
#define H_   2048
#define I_   1408
#define CAP_ 1024   // 2*T*K/E

// ---------------- bf16 helpers ----------------
typedef __bf16 bf16_t;
typedef bf16_t bf16x16 __attribute__((ext_vector_type(16)));
typedef float  v8f     __attribute__((ext_vector_type(8)));

__device__ __forceinline__ unsigned short f2bf(float f) {
  unsigned int u = __float_as_uint(f);
  u += 0x7FFFu + ((u >> 16) & 1u);      // round-to-nearest-even
  return (unsigned short)(u >> 16);
}
__device__ __forceinline__ float bf2f(unsigned short s) {
  return __uint_as_float(((unsigned int)s) << 16);
}
__device__ __forceinline__ uint4 pack8(const float* f) {
  uint4 r;
  r.x = (unsigned)f2bf(f[0]) | ((unsigned)f2bf(f[1]) << 16);
  r.y = (unsigned)f2bf(f[2]) | ((unsigned)f2bf(f[3]) << 16);
  r.z = (unsigned)f2bf(f[4]) | ((unsigned)f2bf(f[5]) << 16);
  r.w = (unsigned)f2bf(f[6]) | ((unsigned)f2bf(f[7]) << 16);
  return r;
}

union Frag { bf16x16 v; uint4 q[2]; unsigned short s[16]; };

// LDS byte offset of a __shared__ object (generic LDS addr truncates to offset)
#define LDS_OFF(p) ((unsigned)(unsigned long long)(p))

// ---------------- routing ----------------
__global__ void zero_counts_k(int* cnt) {
  if (threadIdx.x < E_) cnt[threadIdx.x] = 0;
}

__global__ void route_k(const int* __restrict__ topk_idx,
                        int* __restrict__ cnt,
                        int* __restrict__ tokbuf,     // [E, CAP] source token per slot
                        int* __restrict__ slotbuf) {  // [T*K] global slot or -1
  int i = blockIdx.x * blockDim.x + threadIdx.x;
  if (i >= T_ * K_) return;
  int e = topk_idx[i];
  int p = atomicAdd(&cnt[e], 1);
  int slot = -1;
  if (p < CAP_) { slot = e * CAP_ + p; tokbuf[slot] = i / K_; }
  slotbuf[i] = slot;
}

// ---------------- GEMM1: xbuf @ gu^T -> SwiGLU -> act (bf16) ----------------
// Block tile: 128 rows (c) x 64 cols (i), gate and up halves; K staged 64 deep.
// 8 waves; wave w owns a 16-row stripe: 4 gate + 4 up 16x16 accum tiles.
__global__ __launch_bounds__(256) void gemm1_k(
    const float* __restrict__ x,        // [T, H]
    const float* __restrict__ gu,       // [E, 2I, H]
    const int*   __restrict__ cnt,
    const int*   __restrict__ tokbuf,   // [E, CAP]
    unsigned short* __restrict__ act) { // [E, CAP, I] bf16
  const int e  = blockIdx.x >> 3;       // CAP/128 = 8 row tiles
  const int ct = blockIdx.x & 7;
  const int rows0 = ct * 128;
  int nc = cnt[e]; if (nc > CAP_) nc = CAP_;
  if (rows0 >= nc) return;
  const int i0 = blockIdx.y * 64;

  __shared__ __align__(16) unsigned short As[128 * 64];   // 16 KB
  __shared__ __align__(16) unsigned short Bg[64 * 64];    //  8 KB, B^T: [col][k]
  __shared__ __align__(16) unsigned short Bu[64 * 64];    //  8 KB
  __shared__ int tokrow[128];

  const int tid = threadIdx.x;
  const int lane = tid & 31, w = tid >> 5;

  if (tid < 128) {
    int r = rows0 + tid;
    tokrow[tid] = (r < nc) ? tokbuf[e * CAP_ + r] : -1;
  }
  __syncthreads();

  v8f accg[4], accu[4];
#pragma unroll
  for (int t = 0; t < 4; ++t) {
    v8f z = {0.f, 0.f, 0.f, 0.f, 0.f, 0.f, 0.f, 0.f};
    accg[t] = z; accu[t] = z;
  }

  const size_t gu_base = (size_t)e * (2 * (size_t)I_) * H_;
  const int m  = lane & 15;              // M (A) / N (B) index
  const int kb = (lane >> 4) * 8;        // K half select per ISA layout

  for (int h0 = 0; h0 < H_; h0 += 64) {
    // ---- stage A: 128 rows x 64 k of gathered x, fp32 -> bf16 ----
    {
      int row = tid >> 1, kk = (tid & 1) * 16;
      int tok = tokrow[row];
#pragma unroll
      for (int half = 0; half < 2; ++half) {
        int k0 = kk + half * 32;
        float f[16];
        if (tok >= 0) {
          const float4* src = (const float4*)(x + (size_t)tok * H_ + h0 + k0);
          float4 a0 = src[0], a1 = src[1], a2 = src[2], a3 = src[3];
          f[0]=a0.x; f[1]=a0.y; f[2]=a0.z; f[3]=a0.w;
          f[4]=a1.x; f[5]=a1.y; f[6]=a1.z; f[7]=a1.w;
          f[8]=a2.x; f[9]=a2.y; f[10]=a2.z; f[11]=a2.w;
          f[12]=a3.x; f[13]=a3.y; f[14]=a3.z; f[15]=a3.w;
        } else {
#pragma unroll
          for (int j = 0; j < 16; ++j) f[j] = 0.f;
        }
        *(uint4*)&As[row * 64 + k0]     = pack8(f);
        *(uint4*)&As[row * 64 + k0 + 8] = pack8(f + 8);
      }
    }
    // ---- stage B (gate and up weight slabs), fp32 -> bf16 ----
    {
      int col = tid >> 2, kk = (tid & 3) * 8;
      const float* sg = gu + gu_base + (size_t)(i0 + col) * H_ + h0 + kk;
      const float* su = gu + gu_base + (size_t)(I_ + i0 + col) * H_ + h0 + kk;
      if (h0 + 64 < H_) {                       // prefetch next K-slab
        __builtin_prefetch(sg + 64, 0, 0);
        __builtin_prefetch(su + 64, 0, 0);
      }
#pragma unroll
      for (int half = 0; half < 2; ++half) {
        int k0 = kk + half * 32;
        float4 b0, b1; float f[8];
        b0 = ((const float4*)(sg + half * 32))[0];
        b1 = ((const float4*)(sg + half * 32))[1];
        f[0]=b0.x; f[1]=b0.y; f[2]=b0.z; f[3]=b0.w; f[4]=b1.x; f[5]=b1.y; f[6]=b1.z; f[7]=b1.w;
        *(uint4*)&Bg[col * 64 + k0] = pack8(f);
        b0 = ((const float4*)(su + half * 32))[0];
        b1 = ((const float4*)(su + half * 32))[1];
        f[0]=b0.x; f[1]=b0.y; f[2]=b0.z; f[3]=b0.w; f[4]=b1.x; f[5]=b1.y; f[6]=b1.z; f[7]=b1.w;
        *(uint4*)&Bu[col * 64 + k0] = pack8(f);
      }
    }
    __syncthreads();

    // ---- compute: 16 WMMAs per wave per stage ----
    const unsigned short* arow = &As[(w * 16 + m) * 64];
#pragma unroll
    for (int c = 0; c < 64; c += 32) {
      Frag a;
      a.q[0] = *(const uint4*)&arow[c + kb];
      a.q[1] = *(const uint4*)&arow[c + 16 + kb];
#pragma unroll
      for (int t = 0; t < 4; ++t) {
        Frag b;
        const unsigned short* brow = &Bg[(t * 16 + m) * 64];
        b.q[0] = *(const uint4*)&brow[c + kb];
        b.q[1] = *(const uint4*)&brow[c + 16 + kb];
        accg[t] = __builtin_amdgcn_wmma_f32_16x16x32_bf16(false, a.v, false, b.v,
                                                          (short)0, accg[t], false, false);
        const unsigned short* urow = &Bu[(t * 16 + m) * 64];
        b.q[0] = *(const uint4*)&urow[c + kb];
        b.q[1] = *(const uint4*)&urow[c + 16 + kb];
        accu[t] = __builtin_amdgcn_wmma_f32_16x16x32_bf16(false, a.v, false, b.v,
                                                          (short)0, accu[t], false, false);
      }
    }
    __syncthreads();
  }

  // ---- SwiGLU + store bf16 act ----
  const int rb = (lane >> 4) * 8;    // C/D layout: row = r + 8*(lane>=16)
  const size_t rowbase = (size_t)e * CAP_ + rows0 + w * 16;
#pragma unroll
  for (int t = 0; t < 4; ++t) {
#pragma unroll
    for (int r = 0; r < 8; ++r) {
      float g = accg[t][r], u = accu[t][r];
      float s = g / (1.f + __expf(-g));
      act[(rowbase + r + rb) * I_ + (i0 + t * 16 + m)] = f2bf(s * u);
    }
  }
}

// ---------------- GEMM2: act @ dn^T -> ybuf (bf16) ----------------
// Block tile: 128 rows (c) x 64 cols (h); K staged 64 deep.
// A operand is already bf16 -> stage via async global->LDS (ASYNCcnt path).
__global__ __launch_bounds__(256) void gemm2_k(
    const unsigned short* __restrict__ act,  // [E, CAP, I] bf16
    const float* __restrict__ dn,            // [E, H, I]
    const int*   __restrict__ cnt,
    unsigned short* __restrict__ ybuf) {     // [E, CAP, H] bf16
  const int e  = blockIdx.x >> 3;
  const int ct = blockIdx.x & 7;
  const int rows0 = ct * 128;
  int nc = cnt[e]; if (nc > CAP_) nc = CAP_;
  if (rows0 >= nc) return;
  const int h0 = blockIdx.y * 64;

  __shared__ __align__(16) unsigned short As[128 * 64];   // 16 KB
  __shared__ __align__(16) unsigned short Bs[64 * 64];    //  8 KB

  const int tid = threadIdx.x, lane = tid & 31, w = tid >> 5;
  const int m  = lane & 15;
  const int kb = (lane >> 4) * 8;

  v8f acc[4];
#pragma unroll
  for (int t = 0; t < 4; ++t) {
    v8f z = {0.f, 0.f, 0.f, 0.f, 0.f, 0.f, 0.f, 0.f};
    acc[t] = z;
  }

  const size_t dn_base  = (size_t)e * H_ * I_;
  const size_t act_base = ((size_t)e * CAP_ + rows0) * I_;

  // per-thread A-staging addresses: each thread copies 64 contiguous bytes
  const int arow_ = tid >> 1;
  const int ak_   = (tid & 1) * 32;                 // shorts
  const unsigned aldsoff = LDS_OFF(&As[arow_ * 64 + ak_]);
  const unsigned short* agsrc0 = act + act_base + (size_t)arow_ * I_ + ak_;

  for (int i0 = 0; i0 < I_; i0 += 64) {
    // ---- stage A: async global -> LDS (bf16 passthrough, no VGPR round-trip)
    {
      const unsigned short* gsrc = agsrc0 + i0;
      asm volatile(
          "global_load_async_to_lds_b128 %0, %1, off\n\t"
          "global_load_async_to_lds_b128 %0, %1, off offset:16\n\t"
          "global_load_async_to_lds_b128 %0, %1, off offset:32\n\t"
          "global_load_async_to_lds_b128 %0, %1, off offset:48"
          :: "v"(aldsoff), "v"(gsrc) : "memory");
    }
    // ---- stage B: dn fp32 -> bf16 ----
    {
      int col = tid >> 2, kk = (tid & 3) * 8;
      const float* s = dn + dn_base + (size_t)(h0 + col) * I_ + i0 + kk;
      if (i0 + 64 < I_) __builtin_prefetch(s + 64, 0, 0);
#pragma unroll
      for (int half = 0; half < 2; ++half) {
        float4 b0 = ((const float4*)(s + half * 32))[0];
        float4 b1 = ((const float4*)(s + half * 32))[1];
        float f[8] = {b0.x, b0.y, b0.z, b0.w, b1.x, b1.y, b1.z, b1.w};
        *(uint4*)&Bs[col * 64 + kk + half * 32] = pack8(f);
      }
    }
    asm volatile("s_wait_asynccnt 0x0" ::: "memory");
    __syncthreads();

    // ---- compute: 8 WMMAs per wave per stage ----
    const unsigned short* arow = &As[(w * 16 + m) * 64];
#pragma unroll
    for (int c = 0; c < 64; c += 32) {
      Frag a;
      a.q[0] = *(const uint4*)&arow[c + kb];
      a.q[1] = *(const uint4*)&arow[c + 16 + kb];
#pragma unroll
      for (int t = 0; t < 4; ++t) {
        Frag b;
        const unsigned short* brow = &Bs[(t * 16 + m) * 64];
        b.q[0] = *(const uint4*)&brow[c + kb];
        b.q[1] = *(const uint4*)&brow[c + 16 + kb];
        acc[t] = __builtin_amdgcn_wmma_f32_16x16x32_bf16(false, a.v, false, b.v,
                                                         (short)0, acc[t], false, false);
      }
    }
    __syncthreads();
  }

  const int rb = (lane >> 4) * 8;
  const size_t rowbase = (size_t)e * CAP_ + rows0 + w * 16;
#pragma unroll
  for (int t = 0; t < 4; ++t) {
#pragma unroll
    for (int r = 0; r < 8; ++r) {
      ybuf[(rowbase + r + rb) * H_ + (h0 + t * 16 + m)] = f2bf(acc[t][r]);
    }
  }
}

// ---------------- combine: deterministic per-token gather ----------------
__global__ __launch_bounds__(256) void combine_k(
    const unsigned short* __restrict__ ybuf,  // [E*CAP, H] bf16
    const float* __restrict__ wts,            // [T, K]
    const int*   __restrict__ slotbuf,        // [T*K]
    float* __restrict__ out) {                // [T, H]
  const int t = blockIdx.x;
  int   sl[K_];
  float wk[K_];
#pragma unroll
  for (int k = 0; k < K_; ++k) {
    sl[k] = slotbuf[t * K_ + k];
    wk[k] = wts[t * K_ + k];
  }
  const int h = threadIdx.x * 8;              // H/8 = 256 = blockDim
  float s[8];
#pragma unroll
  for (int j = 0; j < 8; ++j) s[j] = 0.f;
#pragma unroll
  for (int k = 0; k < K_; ++k) {
    if (sl[k] >= 0) {
      uint4 q = *(const uint4*)(ybuf + (size_t)sl[k] * H_ + h);
      const unsigned* qs = (const unsigned*)&q;
#pragma unroll
      for (int j = 0; j < 4; ++j) {
        s[2*j]   += wk[k] * bf2f((unsigned short)(qs[j] & 0xFFFFu));
        s[2*j+1] += wk[k] * bf2f((unsigned short)(qs[j] >> 16));
      }
    }
  }
  float4* o = (float4*)(out + (size_t)t * H_ + h);
  o[0] = make_float4(s[0], s[1], s[2], s[3]);
  o[1] = make_float4(s[4], s[5], s[6], s[7]);
}

// ---------------- launch ----------------
extern "C" void kernel_launch(void* const* d_in, const int* in_sizes, int n_in,
                              void* d_out, int out_size, void* d_ws, size_t ws_size,
                              hipStream_t stream) {
  const float* x   = (const float*)d_in[0];  // hidden_states [T,H]
  const int*   idx = (const int*)  d_in[1];  // top_k_index   [T,K]
  const float* wts = (const float*)d_in[2];  // top_k_weights [T,K]
  const float* gu  = (const float*)d_in[3];  // gate_up_proj  [E,2I,H]
  const float* dn  = (const float*)d_in[4];  // down_proj     [E,H,I]
  float* out = (float*)d_out;

  // workspace layout (bytes)
  char* ws = (char*)d_ws;
  int* cnt     = (int*)ws;                                  // 256 B slab
  int* tokbuf  = (int*)(ws + 256);                          // E*CAP*4 = 128 KB
  int* slotbuf = (int*)(ws + 256 + (size_t)E_ * CAP_ * 4);  // T*K*4  = 64 KB
  size_t off_act = 256 + (size_t)E_ * CAP_ * 4 + (size_t)T_ * K_ * 4;  // 256B aligned
  unsigned short* act  = (unsigned short*)(ws + off_act);               // E*CAP*I*2 ~ 92 MB
  unsigned short* ybuf = (unsigned short*)(ws + off_act + (size_t)E_ * CAP_ * I_ * 2); // ~134 MB

  zero_counts_k<<<1, 32, 0, stream>>>(cnt);
  route_k<<<(T_ * K_ + 255) / 256, 256, 0, stream>>>(idx, cnt, tokbuf, slotbuf);

  dim3 g1(E_ * (CAP_ / 128), I_ / 64);   // 256 x 22
  gemm1_k<<<g1, 256, 0, stream>>>(x, gu, cnt, tokbuf, act);

  dim3 g2(E_ * (CAP_ / 128), H_ / 64);   // 256 x 32
  gemm2_k<<<g2, 256, 0, stream>>>(act, dn, cnt, ybuf);

  combine_k<<<T_, 256, 0, stream>>>(ybuf, wts, slotbuf, out);
}
